// Mask2Former_67370857005307
// MI455X (gfx1250) — matrix operand order repaired and checked by reading, not compiled, not run on hardware
//
#include <hip/hip_runtime.h>
#include <hip/hip_fp16.h>

// Problem constants (from reference)
#define BQ   8      // batch
#define NQ   100    // queries
#define QPAD 128    // queries padded to 4 WMMA K=32 chunks
#define NC   2      // kept classes
#define MH   128    // mask h
#define MW   128    // mask w
#define OH   512    // output H (4x upsample)
#define OW   512    // output W

#define NCHUNK 4            // QPAD / 32
#define TPW    4            // 16-pixel tiles per wave

typedef __attribute__((ext_vector_type(16))) _Float16 v16h;
typedef __attribute__((ext_vector_type(8)))  float    v8f;

#define LOG2E 1.4426950408889634f

__device__ __forceinline__ float fast_sigmoid(float x) {
  // 1/(1+exp(-x)) via v_exp_f32 + v_rcp_f32 (trans pipe)
  float e = __builtin_amdgcn_exp2f(x * (-LOG2E));
  return __builtin_amdgcn_rcpf(1.0f + e);
}

// ---------------------------------------------------------------------------
// Kernel 1: build A-matrix fragments (class softmax weights) directly in the
// contiguous-K 16x32 A per-lane layout (same K order as B), zero-padded to QPAD:
//   lanes 0-15  hold row M=lane,    K in {0..15}
//   lanes 16-31 hold row M=lane-16, K in {16..31}
// Rows M>=2 (unused classes) are zero; q>=NQ padding is zero.
// aPack layout: [b][chunk][lane][16 halves]  (32 KB total).
// ---------------------------------------------------------------------------
__global__ void k_prepA(const float* __restrict__ cls,
                        _Float16* __restrict__ aPack) {
  int t = blockIdx.x * blockDim.x + threadIdx.x;   // 0 .. BQ*NCHUNK*32-1
  if (t >= BQ * NCHUNK * 32) return;
  const int b    = t >> 7;          // / (NCHUNK*32)
  const int ch   = (t >> 5) & 3;
  const int lane = t & 31;
  const int M    = lane & 15;
  const int hiK  = lane >> 4;

  v16h a;
#pragma unroll
  for (int e = 0; e < 16; ++e) {
    const int kA = e + (hiK ? 16 : 0);
    const int q  = ch * 32 + kA;
    float v = 0.0f;
    if (M < NC && q < NQ) {
      const float* l = cls + ((size_t)b * NQ + q) * (NC + 1);
      float a0 = l[0], a1 = l[1], a2 = l[2];
      float mx = fmaxf(a0, fmaxf(a1, a2));
      float e0 = __builtin_amdgcn_exp2f((a0 - mx) * LOG2E);
      float e1 = __builtin_amdgcn_exp2f((a1 - mx) * LOG2E);
      float e2 = __builtin_amdgcn_exp2f((a2 - mx) * LOG2E);
      float r  = __builtin_amdgcn_rcpf(e0 + e1 + e2);
      v = (M == 0 ? e0 : e1) * r;
    }
    a[e] = (_Float16)v;
  }
  ((v16h*)aPack)[t] = a;   // 2x global_store_b128
}

// ---------------------------------------------------------------------------
// Kernel 2: fused 4x bilinear upsample + sigmoid + WMMA reduction over q.
// One wave = TPW consecutive 16-pixel tiles (64 pixels, one output row slice).
// Per tile:  D[16x16] += A[16x32](weights) x B[32x16](sigmoids), 4 K-chunks,
// f32 accumulate in C/D. B-fragment layout: lane holds column N=lane&15
// (its pixel), K = (lane>>4)*16 + e, packed pairs per VGPR.
// ---------------------------------------------------------------------------
__global__ void __launch_bounds__(256)
k_fused(const float* __restrict__ masks,
        const _Float16* __restrict__ aPack,
        float* __restrict__ out) {
  const int lane  = threadIdx.x & 31;
  const int wave  = threadIdx.x >> 5;
  const int tile0 = (blockIdx.x * 8 + wave) * TPW;   // first 16-pixel tile
  const int tilesPerB = (OH * OW) / 16;              // 16384
  const int b      = tile0 / tilesPerB;
  const int tb     = tile0 % tilesPerB;              // multiple of TPW

  const int M   = lane & 15;   // pixel within tile; also D row index space
  const int hiK = lane >> 4;   // B frag: 0 -> K 0..15, 1 -> K 16..31

  // ---- A fragments: 8x global_load_b128, no masking VALU.
  const v16h* ap = (const v16h*)aPack + (size_t)b * NCHUNK * 32 + lane;
  v16h afrag[NCHUNK];
#pragma unroll
  for (int ch = 0; ch < NCHUNK; ++ch) afrag[ch] = ap[ch * 32];

  // ---- y-direction bilinear setup: constant across the TPW tiles
  // (64-pixel group is row-aligned within OW=512).
  const int p0 = tb * 16 + M;
  const int y  = p0 >> 9;                       // / OW
  float fy = (y + 0.5f) * 0.25f - 0.5f;
  int   iy = (int)floorf(fy);  float wy = fy - (float)iy;
  if (iy < 0)      { iy = 0;      wy = 0.0f; }
  if (iy > MH - 2) { iy = MH - 2; wy = 1.0f; }

  const float* mb  = masks + (size_t)b * NQ * MH * MW;
  float*       ob  = out   + (size_t)b * NC * OH * OW;
  const size_t rowb = (size_t)iy * MW;

#pragma unroll 1
  for (int t = 0; t < TPW; ++t) {
    const int p = p0 + t * 16;
    const int x = p & (OW - 1);

    float fx = (x + 0.5f) * 0.25f - 0.5f;
    int   ix = (int)floorf(fx);  float wx = fx - (float)ix;
    if (ix < 0)      { ix = 0;      wx = 0.0f; }
    if (ix > MW - 2) { ix = MW - 2; wx = 1.0f; }

    const size_t off0 = rowb + ix;
    const size_t off1 = off0 + MW;

    v8f acc = {};
#pragma unroll
    for (int ch = 0; ch < NCHUNK; ++ch) {
      const int qb = ch * 32 + hiK * 16;
      v16h bfrag;
#pragma unroll
      for (int e = 0; e < 16; ++e) {
        const int q = qb + e;
        float v = 0.0f;
        if (q < NQ) {                          // skip zero-weight padding
          const float* mq = mb + (size_t)q * (MH * MW);
          float v00 = mq[off0], v01 = mq[off0 + 1];
          float v10 = mq[off1], v11 = mq[off1 + 1];
          float top = v00 + wx * (v01 - v00);
          float bot = v10 + wx * (v11 - v10);
          v = fast_sigmoid(top + wy * (bot - top));
        }
        bfrag[e] = (_Float16)v;
      }
      // EXEC all-1s here: no early returns, branches above reconverged.
      acc = __builtin_amdgcn_wmma_f32_16x16x32_f16(
          /*neg_a=*/false, afrag[ch], /*neg_b=*/false, bfrag,
          /*c_mod=*/(short)0, acc, /*reuse_a=*/false, /*reuse_b=*/false);
    }

    // D rows M=0 (class 0), M=1 (class 1) live in VGPR0/1 of lanes 0..15.
    if (lane < 16) {
      ob[p]                   = acc[0];
      ob[(size_t)OH * OW + p] = acc[1];
    }
  }
}

extern "C" void kernel_launch(void* const* d_in, const int* in_sizes, int n_in,
                              void* d_out, int out_size, void* d_ws, size_t ws_size,
                              hipStream_t stream) {
  const float* cls   = (const float*)d_in[0];  // [8,100,3] f32
  const float* masks = (const float*)d_in[1];  // [8,100,128,128] f32
  float*       out   = (float*)d_out;          // [8,2,512,512] f32
  _Float16*    aPack = (_Float16*)d_ws;        // 8*4*32*16 halves = 32 KB

  k_prepA<<<(BQ * NCHUNK * 32 + 255) / 256, 256, 0, stream>>>(cls, aPack);

  const int tiles  = BQ * ((OH * OW) / 16);        // 131072 wave-tiles
  const int blocks = tiles / (8 * TPW);            // 8 waves x TPW tiles each
  k_fused<<<blocks, 256, 0, stream>>>(masks, aPack, out);
}
